// AttentionDecoder_63058709840000
// MI455X (gfx1250) — matrix operand order repaired and checked by reading, not compiled, not run on hardware
//
#include <hip/hip_runtime.h>
#include <hip/hip_bf16.h>
#include <math.h>

#define V_ 50257
#define E_ 1024
#define H_ 1024
#define B_ 32
#define S_ 1024

typedef _Float16 half16 __attribute__((ext_vector_type(16)));
typedef float    float8 __attribute__((ext_vector_type(8)));

union H16U { half16 v; uint4 q[2]; };
union H8U  { _Float16 h[8]; uint4 q; };

__device__ __forceinline__ float8 f8zero() {
    float8 z;
#pragma unroll
    for (int i = 0; i < 8; ++i) z[i] = 0.0f;
    return z;
}

__device__ __forceinline__ half16 pack16(float4 a, float4 b, float4 c, float4 d) {
    half16 h;
    h[0]  = (_Float16)a.x; h[1]  = (_Float16)a.y; h[2]  = (_Float16)a.z; h[3]  = (_Float16)a.w;
    h[4]  = (_Float16)b.x; h[5]  = (_Float16)b.y; h[6]  = (_Float16)b.z; h[7]  = (_Float16)b.w;
    h[8]  = (_Float16)c.x; h[9]  = (_Float16)c.y; h[10] = (_Float16)c.z; h[11] = (_Float16)c.w;
    h[12] = (_Float16)d.x; h[13] = (_Float16)d.y; h[14] = (_Float16)d.z; h[15] = (_Float16)d.w;
    return h;
}

// A fragment (16x32 f16): lanes 0-15 row M=lane hold K 0..7 (v0-3) and 16..23 (v4-7);
// lanes 16-31 hold K 8..15 and 24..31.
__device__ __forceinline__ half16 load_a_f32(const float* __restrict__ row, int kk, int koffA) {
    const float4* p0 = reinterpret_cast<const float4*>(row + kk + koffA);
    const float4* p1 = reinterpret_cast<const float4*>(row + kk + 16 + koffA);
    return pack16(p0[0], p0[1], p1[0], p1[1]);
}

// B fragment (32x16 f16): lanes 0-15 col N=lane hold K 0..7 and 16..23; lanes 16-31 hold K 8..15 and 24..31.
__device__ __forceinline__ half16 load_b_f32(const float* __restrict__ row, int kk, int koffB) {
    const float4* p0 = reinterpret_cast<const float4*>(row + kk + koffB);
    const float4* p1 = reinterpret_cast<const float4*>(row + kk + 16 + koffB);
    return pack16(p0[0], p0[1], p1[0], p1[1]);
}

__device__ __forceinline__ float8 wmma16(half16 a, half16 b, float8 c) {
    return __builtin_amdgcn_wmma_f32_16x16x32_f16(false, a, false, b, (short)0, c, false, false);
}

// ---------------------------------------------------------------------------
// Kernel 0: pre-convert enc-half of W_a1 to f16 once (it is re-read by 2048
// workgroups in k_scores; converting here removes all B-side cvt VALU from the
// hot loop and halves its L2 traffic).  Whf[n,k] = (f16) W_a1[n,k], k < H.
// ---------------------------------------------------------------------------
__global__ void __launch_bounds__(256)
k_cvtwh(const float* __restrict__ Wa1, _Float16* __restrict__ Whf) {
    const int i = blockIdx.x * 256 + threadIdx.x;     // 131072 threads, 8 elems each
    const int n = i >> 7, k0 = (i & 127) << 3;
    const float4 f0 = *reinterpret_cast<const float4*>(Wa1 + (size_t)n * (2 * H_) + k0);
    const float4 f1 = *reinterpret_cast<const float4*>(Wa1 + (size_t)n * (2 * H_) + k0 + 4);
    H8U u;
    u.h[0] = (_Float16)f0.x; u.h[1] = (_Float16)f0.y;
    u.h[2] = (_Float16)f0.z; u.h[3] = (_Float16)f0.w;
    u.h[4] = (_Float16)f1.x; u.h[5] = (_Float16)f1.y;
    u.h[6] = (_Float16)f1.z; u.h[7] = (_Float16)f1.w;
    *reinterpret_cast<uint4*>(Whf + (size_t)n * H_ + k0) = u.q;
}

// ---------------------------------------------------------------------------
// Kernel 1: hterm[b,h] = b_a1[h] + sum_k h_prev[b,k] * W_a1[h, H+k]
// ---------------------------------------------------------------------------
__global__ void __launch_bounds__(256)
k_hterm(const float* __restrict__ h_prev, const float* __restrict__ Wa1,
        const float* __restrict__ ba1, float* __restrict__ hterm) {
    int i = blockIdx.x * 256 + threadIdx.x;           // B_*H_ = 32768
    int b = i >> 10, h = i & 1023;
    const float* hp = h_prev + (size_t)b * H_;
    const float* w  = Wa1 + (size_t)h * (2 * H_) + H_;
    float acc = ba1[h];
#pragma unroll 4
    for (int k = 0; k < H_; k += 4) {
        float4 a = *reinterpret_cast<const float4*>(hp + k);
        float4 ww = *reinterpret_cast<const float4*>(w + k);
        acc += a.x * ww.x + a.y * ww.y + a.z * ww.z + a.w * ww.w;
    }
    hterm[i] = acc;
}

// ---------------------------------------------------------------------------
// Kernel 2: fused scores[b,s] = b_a2 + W_a2 . tanh(enc[b,s,:]@Wa1_enc^T + hterm[b,:])
// 1 block per (s-tile of 16, b). 8 waves x 8 n-tiles = 64 H-tiles. WMMA f16->f32.
// B operand pre-converted to f16 (Whf) -> pure b128 loads in the hot loop.
// ---------------------------------------------------------------------------
__global__ void __launch_bounds__(256)
k_scores(const float* __restrict__ enc, const _Float16* __restrict__ Whf,
         const float* __restrict__ Wa2, const float* __restrict__ ba2,
         const float* __restrict__ hterm, float* __restrict__ scores) {
    __shared__ _Float16 As[16 * 1032];     // pad 8 halves: 2064B row stride = 4-bank skew
    __shared__ float part[8][16];
    const int b = blockIdx.y, s0 = blockIdx.x * 16, tid = threadIdx.x;

    // stage A tile (16 x 1024) f32 -> f16 LDS
    for (int i = tid; i < 16 * 256; i += 256) {
        int row = i >> 8, c4 = (i & 255) << 2;
        float4 f = *reinterpret_cast<const float4*>(
            enc + ((size_t)(b * S_ + s0 + row)) * H_ + c4);
        _Float16* d = &As[row * 1032 + c4];
        d[0] = (_Float16)f.x; d[1] = (_Float16)f.y;
        d[2] = (_Float16)f.z; d[3] = (_Float16)f.w;
    }
    __syncthreads();

    const int wave = tid >> 5, lane = tid & 31;
    const int koffA = (lane & 16) ? 8 : 0;
    const int koffB = (lane & 16) ? 8 : 0;
    const int nlo = lane & 15;

    float8 acc[8];
#pragma unroll
    for (int t = 0; t < 8; ++t) acc[t] = f8zero();

    for (int kk = 0; kk < H_; kk += 32) {
        H16U a;
        const _Float16* ab = &As[nlo * 1032 + kk + koffA];
        a.q[0] = *reinterpret_cast<const uint4*>(ab);
        a.q[1] = *reinterpret_cast<const uint4*>(ab + 16);
#pragma unroll
        for (int t = 0; t < 8; ++t) {
            const int n = (wave * 8 + t) * 16 + nlo;
            const _Float16* wb = Whf + (size_t)n * H_ + kk + koffB;
            H16U bu;
            bu.q[0] = *reinterpret_cast<const uint4*>(wb);
            bu.q[1] = *reinterpret_cast<const uint4*>(wb + 16);
            acc[t] = wmma16(a.v, bu.v, acc[t]);
        }
    }

    // epilogue: tanh, dot with W_a2, reduce over N
    float p8[8];
#pragma unroll
    for (int r = 0; r < 8; ++r) p8[r] = 0.0f;
#pragma unroll
    for (int t = 0; t < 8; ++t) {
        const int n = (wave * 8 + t) * 16 + nlo;
        const float ht = hterm[(size_t)b * H_ + n];
        const float w2 = Wa2[n];
#pragma unroll
        for (int r = 0; r < 8; ++r)
            p8[r] += tanhf(acc[t][r] + ht) * w2;
    }
    // reduce across the 16-lane N group (rows stay with lane half)
#pragma unroll
    for (int m = 8; m >= 1; m >>= 1)
#pragma unroll
        for (int r = 0; r < 8; ++r)
            p8[r] += __shfl_xor(p8[r], m, 32);

    if ((lane & 15) == 0) {
        const int rbase = (lane >> 4) * 8;   // lane0: rows 0..7, lane16: rows 8..15
#pragma unroll
        for (int r = 0; r < 8; ++r) part[wave][rbase + r] = p8[r];
    }
    __syncthreads();
    if (tid < 16) {
        float s = ba2[0];
#pragma unroll
        for (int w = 0; w < 8; ++w) s += part[w][tid];
        scores[(size_t)b * S_ + s0 + tid] = s;
    }
}

// ---------------------------------------------------------------------------
// Kernel 3: softmax over S per batch row
// ---------------------------------------------------------------------------
__global__ void __launch_bounds__(256)
k_softmax(const float* __restrict__ scores, float* __restrict__ attn) {
    __shared__ float red[256];
    const int b = blockIdx.x, tid = threadIdx.x;
    const float* sr = scores + (size_t)b * S_;
    float4 v = *reinterpret_cast<const float4*>(sr + tid * 4);
    float mx = fmaxf(fmaxf(v.x, v.y), fmaxf(v.z, v.w));
    red[tid] = mx; __syncthreads();
    for (int o = 128; o > 0; o >>= 1) {
        if (tid < o) red[tid] = fmaxf(red[tid], red[tid + o]);
        __syncthreads();
    }
    mx = red[0]; __syncthreads();
    float4 e;
    e.x = expf(v.x - mx); e.y = expf(v.y - mx);
    e.z = expf(v.z - mx); e.w = expf(v.w - mx);
    red[tid] = e.x + e.y + e.z + e.w; __syncthreads();
    for (int o = 128; o > 0; o >>= 1) {
        if (tid < o) red[tid] += red[tid + o];
        __syncthreads();
    }
    const float inv = 1.0f / red[0];
    float4 o4; o4.x = e.x * inv; o4.y = e.y * inv; o4.z = e.z * inv; o4.w = e.w * inv;
    *reinterpret_cast<float4*>(attn + (size_t)b * S_ + tid * 4) = o4;
}

// ---------------------------------------------------------------------------
// Kernel 4: ec[b, 0:E] = emb[word[b]];  ec[b, E:E+H] = context = attn . enc
// ---------------------------------------------------------------------------
__global__ void __launch_bounds__(128)
k_context_ec(const float* __restrict__ enc, const float* __restrict__ attn,
             const float* __restrict__ emb, const int* __restrict__ word,
             float* __restrict__ ec) {
    __shared__ float at[S_];
    const int b = blockIdx.x, y = blockIdx.y, tid = threadIdx.x;
    if (y < 8) {
        const int e = y * 128 + tid;
        const int idx = word[b];
        ec[(size_t)b * (2 * H_) + e] = emb[(size_t)idx * E_ + e];
    } else {
        const int h = (y - 8) * 128 + tid;
        for (int i = tid; i < S_; i += 128) at[i] = attn[(size_t)b * S_ + i];
        __syncthreads();
        float acc = 0.0f;
        const float* ep = enc + (size_t)b * S_ * H_ + h;
#pragma unroll 4
        for (int s = 0; s < S_; ++s) acc += at[s] * ep[(size_t)s * H_];
        ec[(size_t)b * (2 * H_) + H_ + h] = acc;
    }
}

// ---------------------------------------------------------------------------
// Generic one-wave 32xK @ KxN WMMA tile: C[0:32, n0:n0+16]
// ---------------------------------------------------------------------------
__device__ __forceinline__ void gemm32_tile(
    const float* __restrict__ A, int lda,
    const float* __restrict__ W, int ldw, int K, int n0,
    const float* __restrict__ bias, float* __restrict__ out, int ldo,
    int ncols, bool relu, int lane) {
    const int koffA = (lane & 16) ? 8 : 0;
    const int koffB = (lane & 16) ? 8 : 0;
    const int nlo = lane & 15;
    const int col = n0 + nlo;
    const int wcol = (col < ncols) ? col : (ncols - 1);
    const float* a0row = A + (size_t)nlo * lda;
    const float* a1row = A + (size_t)(nlo + 16) * lda;
    const float* wrow  = W + (size_t)wcol * ldw;
    float8 acc0 = f8zero(), acc1 = f8zero();
    for (int kk = 0; kk < K; kk += 32) {
        if (kk + 128 < K) __builtin_prefetch(wrow + kk + 128, 0, 3);
        half16 bf = load_b_f32(wrow, kk, koffB);
        half16 a0 = load_a_f32(a0row, kk, koffA);
        half16 a1 = load_a_f32(a1row, kk, koffA);
        acc0 = wmma16(a0, bf, acc0);
        acc1 = wmma16(a1, bf, acc1);
    }
    if (col < ncols) {
        const float bb = bias ? bias[col] : 0.0f;
        const int rbase = (lane & 16) ? 8 : 0;
#pragma unroll
        for (int r = 0; r < 8; ++r) {
            float v0 = acc0[r] + bb;
            float v1 = acc1[r] + bb;
            if (relu) { v0 = fmaxf(v0, 0.0f); v1 = fmaxf(v1, 0.0f); }
            out[(size_t)(rbase + r) * ldo + col]      = v0;
            out[(size_t)(16 + rbase + r) * ldo + col] = v1;
        }
    }
}

// Kernel 5: x = relu(ec @ W_c^T + b_c)      [32 x 1024], K=2048
__global__ void __launch_bounds__(32)
k_fc_c(const float* __restrict__ ec, const float* __restrict__ Wc,
       const float* __restrict__ bc, float* __restrict__ x) {
    gemm32_tile(ec, 2 * H_, Wc, 2 * H_, 2 * H_, blockIdx.x * 16,
                bc, x, H_, H_, true, threadIdx.x & 31);
}

// Kernel 6: gi = x @ W_ih^T + b_ih ; gh = h_prev @ W_hh^T + b_hh   [32 x 3072]
__global__ void __launch_bounds__(32)
k_gates(const float* __restrict__ x, const float* __restrict__ h_prev,
        const float* __restrict__ Wih, const float* __restrict__ Whh,
        const float* __restrict__ bih, const float* __restrict__ bhh,
        float* __restrict__ gi, float* __restrict__ gh) {
    const int z = blockIdx.y;
    gemm32_tile(z ? h_prev : x, H_, z ? Whh : Wih, H_, H_, blockIdx.x * 16,
                z ? bhh : bih, z ? gh : gi, 3 * H_, 3 * H_, false, threadIdx.x & 31);
}

// Kernel 7: GRU combine -> h_new (to d_out tail) and relu(h_new) (to ws)
__global__ void __launch_bounds__(256)
k_gru(const float* __restrict__ gi, const float* __restrict__ gh,
      const float* __restrict__ h_prev, float* __restrict__ hnew_out,
      float* __restrict__ arelu) {
    const int i = blockIdx.x * 256 + threadIdx.x;   // 32768
    const int b = i >> 10, h = i & 1023;
    const size_t o = (size_t)b * (3 * H_) + h;
    const float ir = gi[o], iz = gi[o + H_], in_ = gi[o + 2 * H_];
    const float hr = gh[o], hz = gh[o + H_], hn = gh[o + 2 * H_];
    const float r = 1.0f / (1.0f + expf(-(ir + hr)));
    const float z = 1.0f / (1.0f + expf(-(iz + hz)));
    const float n = tanhf(in_ + r * hn);
    const float hp = h_prev[i];
    const float hnew = (1.0f - z) * n + z * hp;
    hnew_out[i] = hnew;
    arelu[i] = fmaxf(hnew, 0.0f);
}

// Kernel 8: logits = relu(h_new) @ W_fc^T + b_fc    [32 x 50257]
__global__ void __launch_bounds__(32)
k_logits(const float* __restrict__ a, const float* __restrict__ Wfc,
         const float* __restrict__ bfc, float* __restrict__ out) {
    gemm32_tile(a, H_, Wfc, H_, H_, blockIdx.x * 16,
                bfc, out, V_, V_, false, threadIdx.x & 31);
}

extern "C" void kernel_launch(void* const* d_in, const int* in_sizes, int n_in,
                              void* d_out, int out_size, void* d_ws, size_t ws_size,
                              hipStream_t stream) {
    const int*   word   = (const int*)d_in[0];
    const float* hidden = (const float*)d_in[1];
    const float* enc    = (const float*)d_in[2];
    const float* emb    = (const float*)d_in[3];
    const float* Wa1    = (const float*)d_in[4];
    const float* ba1    = (const float*)d_in[5];
    const float* Wa2    = (const float*)d_in[6];
    const float* ba2    = (const float*)d_in[7];
    const float* Wc     = (const float*)d_in[8];
    const float* bc     = (const float*)d_in[9];
    const float* Wih    = (const float*)d_in[10];
    const float* Whh    = (const float*)d_in[11];
    const float* bih    = (const float*)d_in[12];
    const float* bhh    = (const float*)d_in[13];
    const float* Wfc    = (const float*)d_in[14];
    const float* bfc    = (const float*)d_in[15];
    float* out = (float*)d_out;

    float* ws    = (float*)d_ws;
    _Float16* Whf = (_Float16*)ws;  ws += (H_ * H_) / 2;   // 2 MB f16 W_a1 enc-half
    float* hterm = ws;  ws += B_ * H_;
    float* scs   = ws;  ws += B_ * S_;
    float* attn  = ws;  ws += B_ * S_;
    float* ec    = ws;  ws += B_ * 2 * H_;
    float* x     = ws;  ws += B_ * H_;
    float* gi    = ws;  ws += B_ * 3 * H_;
    float* gh    = ws;  ws += B_ * 3 * H_;
    float* ar    = ws;  ws += B_ * H_;

    k_cvtwh     <<<dim3((H_ * H_) / (8 * 256)), 256, 0, stream>>>(Wa1, Whf);
    k_hterm     <<<dim3((B_ * H_) / 256), 256, 0, stream>>>(hidden, Wa1, ba1, hterm);
    k_scores    <<<dim3(S_ / 16, B_),     256, 0, stream>>>(enc, Whf, Wa2, ba2, hterm, scs);
    k_softmax   <<<dim3(B_),              256, 0, stream>>>(scs, attn);
    k_context_ec<<<dim3(B_, 16),          128, 0, stream>>>(enc, attn, emb, word, ec);
    k_fc_c      <<<dim3(H_ / 16),          32, 0, stream>>>(ec, Wc, bc, x);
    k_gates     <<<dim3(3 * H_ / 16, 2),   32, 0, stream>>>(x, hidden, Wih, Whh, bih, bhh, gi, gh);
    k_gru       <<<dim3((B_ * H_) / 256), 256, 0, stream>>>(gi, gh, hidden, out + (size_t)B_ * V_, ar);
    k_logits    <<<dim3((V_ + 15) / 16),   32, 0, stream>>>(ar, Wfc, bfc, out);
}
